// SimpleRNN_65171833749534
// MI455X (gfx1250) — matrix operand order repaired and checked by reading, not compiled, not run on hardware
//
#include <hip/hip_runtime.h>

typedef _Float16 h8   __attribute__((ext_vector_type(8)));
typedef _Float16 v16h __attribute__((ext_vector_type(16)));
typedef float    f8   __attribute__((ext_vector_type(8)));
typedef float    v8f  __attribute__((ext_vector_type(8)));

#define RNN_B 64
#define RNN_S 512
#define RNN_I 128
#define RNN_H 1024
#define RNN_O 256
#define HSTR  1032   // padded LDS row stride (f16 elems)

union AB { v16h v; h8 h[2]; };

static __device__ __forceinline__ v8f wmma_f16(const AB& a, const AB& b, v8f c) {
  // v_wmma_f32_16x16x32_f16  D = A*B + C, f32 accumulate
  return __builtin_amdgcn_wmma_f32_16x16x32_f16(false, a.v, false, b.v, (short)0, c, false, false);
}

// Non-temporal B-fragment load (stream, zero reuse -> TH=NT, keep WGP$ clean).
static __device__ __forceinline__ void load_bfrag_nt(AB& bf, const _Float16* base,
                                                     size_t tile, int lane) {
  const h8* bp = (const h8*)base + (tile * 64 + lane * 2);
  bf.h[0] = __builtin_nontemporal_load(bp);
  bf.h[1] = __builtin_nontemporal_load(bp + 1);
}

// Pack a row-major weight W[N][K] (f32) into WMMA-B-ready f16 fragments.
// Tile (nt,kt): B[k][n] = W[n][k]; blob dst[(nt*KT+kt)*512 + L*16 + i] with
// n = nt*16 + L%16, K = kt*32 + 8*(L/16) + i%8 + 16*(i/8)  -> 32 contiguous bytes per lane.
__global__ void pack_wmma_b(const float* __restrict__ W, _Float16* __restrict__ dst,
                            int N, int K) {
  int e    = blockIdx.x * blockDim.x + threadIdx.x;
  int i    = e & 15;
  int L    = (e >> 4) & 31;
  int tile = e >> 9;
  int KT   = K >> 5;
  int kt   = tile % KT;
  int nt   = tile / KT;
  int n = nt * 16 + (L & 15);
  int k = kt * 32 + ((L >> 4) << 3) + (i & 7) + ((i >> 3) << 4);
  dst[e] = (_Float16)W[(size_t)n * K + k];
}

// xproj fragments: for timestep s and batch-tile bt (rows b = bt*16..+15),
// blob xpf[((s*4+bt)*64 + nt)*256 + lane*8 + r] = f16 of C-fragment element
// (M = r + 8*(lane/16) -> b row, N = lane%16 within tile nt). This is exactly the
// WMMA C/D per-lane layout, so the scan initializes accumulators with one b128 load.
__global__ __launch_bounds__(256) void xproj_gemm(
    const float* __restrict__ xs, const _Float16* __restrict__ Bih,
    const float* __restrict__ b_ih, const float* __restrict__ b_hh,
    _Float16* __restrict__ xpf) {
  int mt = blockIdx.x;            // 2048 = 512 s * 4 bt
  int s  = mt >> 2;
  int bt = mt & 3;
  int wave = threadIdx.x >> 5;    // 8 waves
  int lane = threadIdx.x & 31;
  int mr = lane & 15, hi = lane >> 4, nlane = lane & 15;

  // A row: xs[b = bt*16+mr][s][:]  (K = 128 contiguous f32)
  const float* arow = xs + ((size_t)(bt * 16 + mr) * RNN_S + s) * RNN_I;

  v8f c[8];
#pragma unroll
  for (int j = 0; j < 8; ++j) {
    int n = (wave * 8 + j) * 16 + nlane;
    float bias = b_ih[n] + b_hh[n];
#pragma unroll
    for (int r = 0; r < 8; ++r) c[j][r] = bias;
  }

#pragma unroll
  for (int kt = 0; kt < 4; ++kt) {
    AB a;
    f8 lo = *(const f8*)(arow + kt * 32 + 8 * hi);
    f8 hv = *(const f8*)(arow + kt * 32 + 8 * hi + 16);
    a.h[0] = __builtin_convertvector(lo, h8);
    a.h[1] = __builtin_convertvector(hv, h8);
#pragma unroll
    for (int j = 0; j < 8; ++j) {
      int nt = wave * 8 + j;
      AB bf;
      load_bfrag_nt(bf, Bih, (size_t)(nt * 4 + kt), lane);
      c[j] = wmma_f16(a, bf, c[j]);
    }
  }

  // Store each tile as a C-fragment blob: one coalesced non-temporal 16B store/lane.
#pragma unroll
  for (int j = 0; j < 8; ++j) {
    int nt = wave * 8 + j;
    h8* dst = (h8*)(xpf + (((size_t)(s * 4 + bt) * 64 + nt) * 256 + lane * 8));
    __builtin_nontemporal_store(__builtin_convertvector(c[j], h8), dst);
  }
}

// Sequential scan. 4 workgroups, each owns one 16-row batch tile and all 1024
// hidden columns -> no cross-WG communication, one barrier per timestep.
// 8 waves x 8 N-tiles; h double-buffered f16 in LDS; B/xpf streamed non-temporal.
__global__ __launch_bounds__(256) void rnn_scan(
    const _Float16* __restrict__ xpf, const _Float16* __restrict__ Bhh,
    const _Float16* __restrict__ Bho, const float* __restrict__ b_ho,
    float* __restrict__ out) {
  extern __shared__ _Float16 hb[];              // 2 * 16 * HSTR f16
  int bt = blockIdx.x;
  int b0 = bt * 16;
  int wave = threadIdx.x >> 5;                  // 8
  int lane = threadIdx.x & 31;
  int mr = lane & 15, hi = lane >> 4, nlane = lane & 15;

  for (int i = threadIdx.x; i < 16 * HSTR; i += 256) hb[i] = (_Float16)0.f;  // h0 = 0
  __syncthreads();

  int cur = 0;
  for (int t = 0; t < RNN_S; ++t) {
    const _Float16* hcur = hb + cur * (16 * HSTR);

    v8f c[8];
#pragma unroll
    for (int g = 0; g < 8; ++g) {               // accumulator init: 1 NT b128 + cvt
      int nt = wave * 8 + g;
      const h8* xb = (const h8*)(xpf + (((size_t)(t * 4 + bt) * 64 + nt) * 256 + lane * 8));
      c[g] = __builtin_convertvector(__builtin_nontemporal_load(xb), f8);
    }

    for (int kt = 0; kt < 32; ++kt) {
      AB a;                                     // A frag: h rows from LDS
      const _Float16* ap = hcur + mr * HSTR + kt * 32 + 8 * hi;
      a.h[0] = *(const h8*)ap;
      a.h[1] = *(const h8*)(ap + 16);
#pragma unroll
      for (int g = 0; g < 8; ++g) {             // 8 independent WMMA chains
        int nt = wave * 8 + g;
        AB bf;
        load_bfrag_nt(bf, Bhh, (size_t)(nt * 32 + kt), lane);
        c[g] = wmma_f16(a, bf, c[g]);
      }
    }

    _Float16* hnxt = hb + (cur ^ 1) * (16 * HSTR);
#pragma unroll
    for (int g = 0; g < 8; ++g) {
      int n = (wave * 8 + g) * 16 + nlane;
#pragma unroll
      for (int r = 0; r < 8; ++r) {
        float v = c[g][r];
        hnxt[(size_t)(r + 8 * hi) * HSTR + n] = (_Float16)(v > 0.f ? v : 0.f);  // relu
      }
    }
    __syncthreads();
    cur ^= 1;
  }

  // out[b0..b0+15, :] = h_final @ W_ho^T + b_ho   (16 N-tiles, 2 per wave)
  {
    const _Float16* hcur = hb + cur * (16 * HSTR);
#pragma unroll
    for (int q = 0; q < 2; ++q) {
      int nt = wave * 2 + q;
      float bias = b_ho[nt * 16 + nlane];
      v8f c;
#pragma unroll
      for (int r = 0; r < 8; ++r) c[r] = bias;
      for (int kt = 0; kt < 32; ++kt) {
        AB a;
        const _Float16* ap = hcur + mr * HSTR + kt * 32 + 8 * hi;
        a.h[0] = *(const h8*)ap;
        a.h[1] = *(const h8*)(ap + 16);
        AB bf;
        load_bfrag_nt(bf, Bho, (size_t)(nt * 32 + kt), lane);
        c = wmma_f16(a, bf, c);
      }
#pragma unroll
      for (int r = 0; r < 8; ++r)
        __builtin_nontemporal_store(
            c[r], &out[(size_t)(b0 + r + 8 * hi) * RNN_O + nt * 16 + nlane]);
    }
  }
}

extern "C" void kernel_launch(void* const* d_in, const int* in_sizes, int n_in,
                              void* d_out, int out_size, void* d_ws, size_t ws_size,
                              hipStream_t stream) {
  const float* xs   = (const float*)d_in[0];
  const float* W_ih = (const float*)d_in[1];
  const float* b_ih = (const float*)d_in[2];
  const float* W_hh = (const float*)d_in[3];
  const float* b_hh = (const float*)d_in[4];
  const float* W_ho = (const float*)d_in[5];
  const float* b_ho = (const float*)d_in[6];
  float* out = (float*)d_out;

  // Workspace layout (f16 elems): packed weights + fragment-layout xproj (~66.8 MB).
  _Float16* ws    = (_Float16*)d_ws;
  _Float16* Bhh   = ws;                     // 64nt * 32kt * 512 = 1,048,576
  _Float16* Bho   = Bhh + 1048576;          // 16nt * 32kt * 512 =   262,144
  _Float16* Bih   = Bho + 262144;           // 64nt *  4kt * 512 =   131,072
  _Float16* xpf   = Bih + 131072;           // 512s * 4bt * 64nt * 256 = 33,554,432

  pack_wmma_b<<<4096, 256, 0, stream>>>(W_hh, Bhh, RNN_H, RNN_H);
  pack_wmma_b<<<1024, 256, 0, stream>>>(W_ho, Bho, RNN_O, RNN_H);
  pack_wmma_b<<< 512, 256, 0, stream>>>(W_ih, Bih, RNN_H, RNN_I);
  xproj_gemm<<<2048, 256, 0, stream>>>(xs, Bih, b_ih, b_hh, xpf);
  rnn_scan<<<4, 256, 2 * 16 * HSTR * (int)sizeof(_Float16), stream>>>(
      xpf, Bhh, Bho, b_ho, out);
}
